// ChannelAttention_10591389352163
// MI455X (gfx1250) — matrix operand order repaired and checked by reading, not compiled, not run on hardware
//
#include <hip/hip_runtime.h>
#include <hip/hip_bf16.h>

typedef __attribute__((ext_vector_type(16))) _Float16 v16h;
typedef __attribute__((ext_vector_type(8)))  float    v8f;

#define B_  32
#define C_  512
#define HW  6400      // 80*80
#define NSORT 8192    // pad to pow2
#define NBIN 1280     // HW/5
#define SROW 3584     // 7*C_

#if defined(__gfx1250__) && __has_builtin(__builtin_amdgcn_global_load_async_to_lds_b128) && __has_builtin(__builtin_amdgcn_s_wait_asynccnt)
#define USE_ASYNC_LDS 1
#else
#define USE_ASYNC_LDS 0
#endif

#if USE_ASYNC_LDS
typedef int v4i_vs __attribute__((vector_size(16)));
typedef __attribute__((address_space(1))) v4i_vs* g_v4i_ptr;   // global (AS1)
typedef __attribute__((address_space(3))) v4i_vs* l_v4i_ptr;   // LDS (AS3)
typedef const __attribute__((address_space(1))) char* gas_ptr;
typedef __attribute__((address_space(3))) char*       las_ptr;
#endif

// ---------------------------------------------------------------------------
// Kernel 1: per-(b,c) row -> sort in LDS (bitonic), emit all 7 stats:
//   stacked[b][0][c]=avg, [1]=max, [2..6]=quintile means of sorted row
// ---------------------------------------------------------------------------
__global__ __launch_bounds__(256) void k_sortstats(const float* __restrict__ x,
                                                   float* __restrict__ stacked) {
  __shared__ float data[NSORT];
  __shared__ float red[256];
  __shared__ float bsum[5];
  const int tid = threadIdx.x;
  const int row = blockIdx.x;                       // b*C_ + c
  const float* __restrict__ p = x + (size_t)row * HW;

#if USE_ASYNC_LDS
  {
    // 6400 floats = 1600 x B128 chunks, DMA'd straight into LDS (ASYNCcnt).
    gas_ptr gp = (gas_ptr)p;
    las_ptr lp = (las_ptr)data;
    for (int i = tid; i < HW / 4; i += 256)
      __builtin_amdgcn_global_load_async_to_lds_b128(
          (g_v4i_ptr)(gp + (size_t)i * 16),
          (l_v4i_ptr)(lp + i * 16), 0, 0);
    for (int i = HW + tid; i < NSORT; i += 256) data[i] = __builtin_inff();
    __builtin_amdgcn_s_wait_asynccnt(0);
  }
#else
  for (int i = tid; i < NSORT; i += 256)
    data[i] = (i < HW) ? p[i] : __builtin_inff();
#endif
  __syncthreads();

  // bitonic sort, ascending
  for (int kk = 2; kk <= NSORT; kk <<= 1) {
    for (int j = kk >> 1; j > 0; j >>= 1) {
      for (int i = tid; i < NSORT; i += 256) {
        const int ixj = i ^ j;
        if (ixj > i) {
          const float va = data[i], vb = data[ixj];
          const bool asc = ((i & kk) == 0);
          if (asc ? (va > vb) : (va < vb)) { data[i] = vb; data[ixj] = va; }
        }
      }
      __syncthreads();
    }
  }

  // quintile sums (contiguous 1280-element segments of sorted data)
  for (int bin = 0; bin < 5; ++bin) {
    float s = 0.f;
    const int base = bin * NBIN;
    for (int i = base + tid; i < base + NBIN; i += 256) s += data[i];
    red[tid] = s;
    __syncthreads();
    for (int off = 128; off > 0; off >>= 1) {
      if (tid < off) red[tid] += red[tid + off];
      __syncthreads();
    }
    if (tid == 0) bsum[bin] = red[0];
    __syncthreads();
  }

  if (tid == 0) {
    const float total = bsum[0] + bsum[1] + bsum[2] + bsum[3] + bsum[4];
    const int b = row >> 9, c = row & (C_ - 1);
    float* __restrict__ srow = stacked + b * SROW + c;
    srow[0]          = total * (1.f / (float)HW);   // avg (sum is sort-invariant)
    srow[C_]         = data[HW - 1];                // max (largest real element)
    for (int bin = 0; bin < 5; ++bin)
      srow[(2 + bin) * C_] = bsum[bin] * (1.f / (float)NBIN);
  }
}

// ---------------------------------------------------------------------------
// Kernel 2: WMMA MLP.
//   h = relu(stacked(224x512) @ W1^T(512x32))        [GEMM1, f16 WMMA]
//   mlp = h(224x32) @ W2^T(32x512)                   [GEMM2, f16 WMMA]
//   acc[b][c] += w7[row%7] * mlp[row][c]             [atomic scatter]
// One block = one 16-row M tile (14 blocks), 2 waves per block.
// ---------------------------------------------------------------------------
__global__ __launch_bounds__(64) void k_mlp(const float* __restrict__ stacked,
                                            const float* __restrict__ W1,
                                            const float* __restrict__ W2,
                                            const float* __restrict__ w7,
                                            float* __restrict__ acc) {
  __shared__ _Float16 hbuf[16][32];                 // relu(h) tile, A-layout source
  const int tid   = threadIdx.x;
  const int wave  = tid >> 5;
  const int lane  = tid & 31;
  const int mtile = blockIdx.x;                     // 0..13
  const int lrow  = lane & 15;
  const int khalf = lane >> 4;                      // 0 or 1

  // ---- GEMM1: this wave computes h cols [wave*16, wave*16+16)
  v8f c1 = {};
  const int m  = mtile * 16 + lrow;                 // A row (0..223)
  const int n1 = wave * 16 + lrow;                  // B col (0..31) -> W1 row
  for (int kk = 0; kk < 16; ++kk) {
    v16h a, bm;
    const int kbA = kk * 32 + khalf * 8;            // A lane-half K groups {0-7,16-23}/{8-15,24-31}
    const int kbB = kk * 32 + khalf * 16;           // B lane-half K groups {0-7,16-23}/{8-15,24-31} (via ka)
#pragma unroll
    for (int e = 0; e < 16; ++e) {
      const int ka = kbA + (e < 8 ? e : e + 8);
      a[e]  = (_Float16)stacked[m * C_ + ka];       // A[m][k]
      bm[e] = (_Float16)W1[n1 * C_ + ka];           // B[k][n] = W1[n][k]
    }
    c1 = __builtin_amdgcn_wmma_f32_16x16x32_f16(false, a, false, bm,
                                                (short)0, c1, false, false);
  }
  // relu + stage D-layout -> plain 16x32 f16 tile in LDS
#pragma unroll
  for (int r = 0; r < 8; ++r) {
    float v = c1[r];
    v = v > 0.f ? v : 0.f;
    hbuf[r + khalf * 8][wave * 16 + lrow] = (_Float16)v;
  }
  __syncthreads();

  // ---- GEMM2: A = hbuf (16x32), loop over 32 N-tiles split across 2 waves
  v16h a2;
#pragma unroll
  for (int e = 0; e < 16; ++e) {
    const int ka = khalf * 8 + (e < 8 ? e : e + 8);
    a2[e] = hbuf[lrow][ka];
  }

  // Hoist div/mod-by-7 and the w7 gather: depends only on r, not on nt.
  int   bidx[8];
  float wv[8];
#pragma unroll
  for (int r = 0; r < 8; ++r) {
    const int grow = mtile * 16 + r + khalf * 8;    // global row 0..223
    bidx[r] = grow / 7;
    wv[r]   = w7[grow % 7];
  }

  for (int nt = wave; nt < 32; nt += 2) {
    v16h bm;
    const int n2 = nt * 16 + lrow;                  // global col 0..511 -> W2 row
#pragma unroll
    for (int e = 0; e < 16; ++e)
      bm[e] = (_Float16)W2[n2 * 32 + khalf * 8 + (e < 8 ? e : e + 8)];  // B[k][n] = W2[n][k]
    v8f c2 = {};
    c2 = __builtin_amdgcn_wmma_f32_16x16x32_f16(false, a2, false, bm,
                                                (short)0, c2, false, false);
    const int col = nt * 16 + lrow;
#pragma unroll
    for (int r = 0; r < 8; ++r)
      atomicAdd(&acc[bidx[r] * C_ + col], c2[r] * wv[r]);
  }
}

// ---------------------------------------------------------------------------
// Kernel 3: out = sigmoid(acc + bias)
// ---------------------------------------------------------------------------
__global__ __launch_bounds__(256) void k_final(const float* __restrict__ acc,
                                               const float* __restrict__ bias,
                                               float* __restrict__ out) {
  const int i = blockIdx.x * 256 + threadIdx.x;
  if (i < B_ * C_) {
    const float v = acc[i] + bias[0];
    out[i] = 1.f / (1.f + __expf(-v));
  }
}

extern "C" void kernel_launch(void* const* d_in, const int* in_sizes, int n_in,
                              void* d_out, int out_size, void* d_ws, size_t ws_size,
                              hipStream_t stream) {
  const float* x  = (const float*)d_in[0];   // (32,512,80,80)
  const float* W1 = (const float*)d_in[1];   // (32,512)
  const float* W2 = (const float*)d_in[2];   // (512,32)
  const float* w7 = (const float*)d_in[3];   // (7,)
  const float* bb = (const float*)d_in[4];   // (1,)
  float* out = (float*)d_out;                // (32,512)

  float* stacked = (float*)d_ws;             // 32*7*512 floats
  float* acc     = stacked + B_ * SROW;      // 32*512 floats

  k_sortstats<<<B_ * C_, 256, 0, stream>>>(x, stacked);
  (void)hipMemsetAsync(acc, 0, (size_t)B_ * C_ * sizeof(float), stream);
  k_mlp<<<14, 64, 0, stream>>>(stacked, W1, W2, w7, acc);
  k_final<<<(B_ * C_ + 255) / 256, 256, 0, stream>>>(acc, bb, out);
}
